// TransformerBlock_2499670966275
// MI455X (gfx1250) — matrix operand order repaired and checked by reading, not compiled, not run on hardware
//
#include <hip/hip_runtime.h>
#include <hip/hip_bf16.h>
#include <math.h>
#include <stdint.h>

// ---------------------------------------------------------------------------
// Types for CDNA5 WMMA (wave32): 16x16x32 bf16 -> f32
// ---------------------------------------------------------------------------
typedef __bf16 bf16x8  __attribute__((ext_vector_type(8)));
typedef __bf16 bf16x16 __attribute__((ext_vector_type(16)));
typedef float  f32x8   __attribute__((ext_vector_type(8)));

static constexpr int T_SEQ = 2048;
static constexpr int C_EMB = 1024;
static constexpr int HEADS = 16;
static constexpr int HDIM  = 64;
static constexpr int FFDIM = 4096;
static constexpr int MROWS = 2 * T_SEQ; // B*T = 4096

__device__ __forceinline__ f32x8 wmma_bf16(bf16x16 a, bf16x16 b, f32x8 c) {
  return __builtin_amdgcn_wmma_f32_16x16x32_bf16(false, a, false, b,
                                                 (short)0, c, false, false);
}

// A-matrix 16x32 bf16 fragment (ISA 7.12.2): lane group g (=lane>>4) holds
// K = [g*8 .. g*8+7] and [g*8+16 .. g*8+23]; row = lane&15 (caller picks row ptr).
__device__ __forceinline__ bf16x16 frag_a(const __bf16* p, int g) {
  bf16x8 lo = *(const bf16x8*)(p + g * 8);
  bf16x8 hi = *(const bf16x8*)(p + g * 8 + 16);
  bf16x16 v;
#pragma unroll
  for (int i = 0; i < 8; ++i) { v[i] = lo[i]; v[8 + i] = hi[i]; }
  return v;
}

// B-matrix 32x16 bf16 fragment ("column-major" source = rows of B^T, K-contig):
// lane group g holds K = [g*8 .. g*8+7] and [g*8+16 .. g*8+23]; column n = lane&15 (caller picks ptr).
__device__ __forceinline__ bf16x16 frag_b(const __bf16* p, int g) {
  bf16x8 lo = *(const bf16x8*)(p + g * 8);
  bf16x8 hi = *(const bf16x8*)(p + g * 8 + 16);
  bf16x16 v;
#pragma unroll
  for (int i = 0; i < 8; ++i) { v[i] = lo[i]; v[8 + i] = hi[i]; }
  return v;
}

// ---------------------------------------------------------------------------
// gfx1250 async global->LDS DMA (tracked by ASYNCcnt).
// IOFFSET is added to BOTH the LDS and global address, so one base copies a
// contiguous 64-byte row with 4 instructions and zero staging VGPRs.
// ---------------------------------------------------------------------------
__device__ __forceinline__ void async_copy_row64(uint32_t lds_addr,
                                                 const void* gaddr) {
  asm volatile(
      "global_load_async_to_lds_b128 %0, %1, off\n\t"
      "global_load_async_to_lds_b128 %0, %1, off offset:16\n\t"
      "global_load_async_to_lds_b128 %0, %1, off offset:32\n\t"
      "global_load_async_to_lds_b128 %0, %1, off offset:48"
      :: "v"(lds_addr), "v"(gaddr) : "memory");
}
__device__ __forceinline__ void wait_async_le8() {
  asm volatile("s_wait_asynccnt 0x8" ::: "memory");
}
__device__ __forceinline__ void wait_async_0() {
  asm volatile("s_wait_asynccnt 0x0" ::: "memory");
}

// LDS flat address -> 32-bit LDS offset (hardware truncates addr[31:0])
__device__ __forceinline__ uint32_t lds_off(const void* p) {
  return (uint32_t)(uintptr_t)p;
}

// ---------------------------------------------------------------------------
// Packing kernels
// ---------------------------------------------------------------------------
__global__ __launch_bounds__(1024) void pack_x_kernel(const float* __restrict__ in,
                                                      __bf16* __restrict__ out,
                                                      int n) {
  int i = blockIdx.x * blockDim.x + threadIdx.x;
  if (i < n) out[i] = (__bf16)in[i];
}

// W [K][N] fp32 row-major  ->  Wt [N][K] bf16 row-major (so GEMM B is K-contig)
__global__ __launch_bounds__(256) void pack_wT_kernel(const float* __restrict__ W,
                                                      __bf16* __restrict__ Wt,
                                                      int K, int N) {
  __shared__ float tile[32][33];
  int n0 = blockIdx.x * 32;
  int k0 = blockIdx.y * 32;
  for (int i = threadIdx.y; i < 32; i += 8)
    tile[i][threadIdx.x] = W[(size_t)(k0 + i) * N + n0 + threadIdx.x];
  __syncthreads();
  for (int i = threadIdx.y; i < 32; i += 8)
    Wt[(size_t)(n0 + i) * K + k0 + threadIdx.x] = (__bf16)tile[threadIdx.x][i];
}

// ---------------------------------------------------------------------------
// Tiled WMMA GEMM:  out[M][N] = A[M][K] * Bt[N][K]^T  (+ epilogue)
//   MODE 0: bf16 out [M][N]
//   MODE 1: bf16 out transposed per head -> Vt[B][H][HDIM][T]
//   MODE 2: f32 out [M][N] = acc + bias[n] + res[m][n]
//   MODE 3: bf16 out [M][N] = gelu(acc + bias[n])
// Block 128 thr (4 waves), tile 128x128, BK=32, wave tile 64x64 (4x4 frags):
// 16 WMMAs per K tile.  Global->LDS via async DMA, double buffered.
// ---------------------------------------------------------------------------
template <int MODE>
__global__ __launch_bounds__(128) void gemm_bf16_kernel(
    const __bf16* __restrict__ A, const __bf16* __restrict__ Bt,
    int M, int N, int K,
    float* __restrict__ outf, __bf16* __restrict__ outb,
    const float* __restrict__ bias, const float* __restrict__ res) {
  // Row stride 48 elems (96 B): each 16-B chunk stays 16-B aligned.
  __shared__ __bf16 As[2][128 * 48];
  __shared__ __bf16 Bs[2][128 * 48];

  const int tid = threadIdx.x;
  const int wid = tid >> 5;
  const int lid = tid & 31;
  const int g = lid >> 4;        // lane group (K-chunk select)
  const int r = lid & 15;        // row / col within fragment
  const int wm = (wid >> 1) * 64;
  const int wn = (wid & 1) * 64;
  const int bm = blockIdx.y * 128;
  const int bn = blockIdx.x * 128;

  f32x8 acc[4][4];
#pragma unroll
  for (int mt = 0; mt < 4; ++mt)
#pragma unroll
    for (int nt = 0; nt < 4; ++nt) {
      f32x8 z = {};
      acc[mt][nt] = z;
    }

  // Each thread owns one 64-byte row slice of each tile.
  const __bf16* aRow = A + (size_t)(bm + tid) * K;
  const __bf16* bRow = Bt + (size_t)(bn + tid) * K;
  const uint32_t ldsA0 = lds_off(&As[0][tid * 48]);
  const uint32_t ldsA1 = lds_off(&As[1][tid * 48]);
  const uint32_t ldsB0 = lds_off(&Bs[0][tid * 48]);
  const uint32_t ldsB1 = lds_off(&Bs[1][tid * 48]);

  const int ntiles = K >> 5;
  async_copy_row64(ldsA0, aRow);
  async_copy_row64(ldsB0, bRow);

  for (int t = 0; t < ntiles; ++t) {
    const int cur = t & 1;
    if (t + 1 < ntiles) {  // prefetch next K tile into the other buffer
      async_copy_row64(cur ? ldsA0 : ldsA1, aRow + (t + 1) * 32);
      async_copy_row64(cur ? ldsB0 : ldsB1, bRow + (t + 1) * 32);
      wait_async_le8();    // tile t's 8 DMAs done; tile t+1 stays in flight
    } else {
      wait_async_0();
    }
    __syncthreads();

    bf16x16 af[4], bfr[4];
#pragma unroll
    for (int mt = 0; mt < 4; ++mt)
      af[mt] = frag_a(&As[cur][(wm + mt * 16 + r) * 48], g);
#pragma unroll
    for (int nt = 0; nt < 4; ++nt)
      bfr[nt] = frag_b(&Bs[cur][(wn + nt * 16 + r) * 48], g);
#pragma unroll
    for (int mt = 0; mt < 4; ++mt)
#pragma unroll
      for (int nt = 0; nt < 4; ++nt)
        acc[mt][nt] = wmma_bf16(af[mt], bfr[nt], acc[mt][nt]);
    __syncthreads();
  }

  // Epilogue.  C/D layout: lane group g holds rows (8g+i), column r.
#pragma unroll
  for (int mt = 0; mt < 4; ++mt) {
#pragma unroll
    for (int nt = 0; nt < 4; ++nt) {
      const int m0 = bm + wm + mt * 16 + g * 8;       // first of 8 consecutive rows
      const int n = bn + wn + nt * 16 + r;            // column
      if (MODE == 0) {
#pragma unroll
        for (int i = 0; i < 8; ++i)
          outb[(size_t)(m0 + i) * N + n] = (__bf16)acc[mt][nt][i];
      } else if (MODE == 1) {
        // V: write Vt[b][h][d][t]; the 8 rows are consecutive t -> one 16-B store
        const int b = m0 >> 11;            // T = 2048
        const int t0 = m0 & (T_SEQ - 1);
        const int h = n >> 6;
        const int d = n & (HDIM - 1);
        bf16x8 pk;
#pragma unroll
        for (int i = 0; i < 8; ++i) pk[i] = (__bf16)acc[mt][nt][i];
        *(bf16x8*)&outb[((((size_t)b * HEADS + h) * HDIM + d) * T_SEQ) + t0] = pk;
      } else if (MODE == 2) {
        const float bv = bias[n];
#pragma unroll
        for (int i = 0; i < 8; ++i) {
          size_t idx = (size_t)(m0 + i) * N + n;
          outf[idx] = acc[mt][nt][i] + bv + res[idx];
        }
      } else {  // MODE 3: exact GELU, bf16 out
        const float bv = bias[n];
#pragma unroll
        for (int i = 0; i < 8; ++i) {
          float v = acc[mt][nt][i] + bv;
          float gl = 0.5f * v * (1.0f + erff(v * 0.70710678118f));
          outb[(size_t)(m0 + i) * N + n] = (__bf16)gl;
        }
      }
    }
  }
}

// ---------------------------------------------------------------------------
// Flash attention: one wave per (b, h, 16-query block). Online softmax,
// QK^T and PV via WMMA, causal mask. Q/K read [B][T][C] bf16; V read
// pre-transposed Vt[B][H][HDIM][T] bf16; ctx written [B][T][C] bf16.
// ---------------------------------------------------------------------------
__global__ __launch_bounds__(32) void attn_kernel(
    const __bf16* __restrict__ Q, const __bf16* __restrict__ Km,
    const __bf16* __restrict__ Vt, __bf16* __restrict__ ctx) {
  __shared__ __bf16 Ps[16 * 48];  // P tile round-trip (C-layout -> A-layout)

  const int lid = threadIdx.x;
  const int g = lid >> 4;
  const int r = lid & 15;
  const int bh = blockIdx.y;
  const int b = bh >> 4;
  const int h = bh & (HEADS - 1);
  const int q0 = blockIdx.x * 16;

  const __bf16* qbase = Q + ((size_t)b * T_SEQ + q0) * C_EMB + h * HDIM;
  const __bf16* kbase = Km + (size_t)b * T_SEQ * C_EMB + h * HDIM;
  const __bf16* vbase = Vt + ((size_t)b * HEADS + h) * HDIM * T_SEQ;

  // Q fragments for d = [0,32) and [32,64)
  bf16x16 qf[2];
  qf[0] = frag_a(qbase + (size_t)r * C_EMB, g);
  qf[1] = frag_a(qbase + (size_t)r * C_EMB + 32, g);

  float mrow[8], lrow[8];
  f32x8 o[4];
#pragma unroll
  for (int i = 0; i < 8; ++i) { mrow[i] = -3.0e38f; lrow[i] = 0.0f; }
#pragma unroll
  for (int dt = 0; dt < 4; ++dt) { f32x8 z = {}; o[dt] = z; }

  for (int kb = 0; kb < q0 + 16; kb += 32) {
    // S = Q * K^T for 32 keys (two 16-col fragments), K-dim = 64 head dims
    f32x8 s[2];
#pragma unroll
    for (int kt = 0; kt < 2; ++kt) {
      f32x8 z = {};
      const __bf16* krow = kbase + (size_t)(kb + kt * 16 + r) * C_EMB;
      z = wmma_bf16(qf[0], frag_b(krow, g), z);
      z = wmma_bf16(qf[1], frag_b(krow + 32, g), z);
      s[kt] = z;
    }

    float p0[8], p1[8], corr[8];
#pragma unroll
    for (int i = 0; i < 8; ++i) {
      const int qrow = q0 + g * 8 + i;
      float s0 = s[0][i] * 0.125f;  // 1/sqrt(64)
      float s1 = s[1][i] * 0.125f;
      if (kb + r > qrow)      s0 = -3.0e38f;   // causal mask
      if (kb + 16 + r > qrow) s1 = -3.0e38f;
      float mx = fmaxf(s0, s1);
#pragma unroll
      for (int off = 1; off < 16; off <<= 1)
        mx = fmaxf(mx, __shfl_xor(mx, off, 32));  // stays within 16-lane half
      const float mnew = fmaxf(mrow[i], mx);
      const float e0 = __expf(s0 - mnew);
      const float e1 = __expf(s1 - mnew);
      float rs = e0 + e1;
#pragma unroll
      for (int off = 1; off < 16; off <<= 1)
        rs += __shfl_xor(rs, off, 32);
      const float cf = __expf(mrow[i] - mnew);
      lrow[i] = lrow[i] * cf + rs;
      mrow[i] = mnew;
      p0[i] = e0; p1[i] = e1; corr[i] = cf;
    }
#pragma unroll
    for (int dt = 0; dt < 4; ++dt)
#pragma unroll
      for (int i = 0; i < 8; ++i) o[dt][i] *= corr[i];

    // C-layout probs -> LDS -> A-layout bf16 fragment
    __syncthreads();
#pragma unroll
    for (int i = 0; i < 8; ++i) {
      Ps[(g * 8 + i) * 48 + r]      = (__bf16)p0[i];
      Ps[(g * 8 + i) * 48 + 16 + r] = (__bf16)p1[i];
    }
    __syncthreads();
    const bf16x16 pa = frag_a(&Ps[r * 48], g);

    // o += P * V   (B frags from Vt: d = dt*16 + r, t contiguous)
#pragma unroll
    for (int dt = 0; dt < 4; ++dt) {
      const __bf16* vrow = vbase + (size_t)(dt * 16 + r) * T_SEQ + kb;
      o[dt] = wmma_bf16(pa, frag_b(vrow, g), o[dt]);
    }
  }

  // Normalize and write ctx [B][T][C] bf16
#pragma unroll
  for (int dt = 0; dt < 4; ++dt)
#pragma unroll
    for (int i = 0; i < 8; ++i) {
      const float val = o[dt][i] / lrow[i];
      ctx[((size_t)b * T_SEQ + q0 + g * 8 + i) * C_EMB + h * HDIM + dt * 16 + r] =
          (__bf16)val;
    }
}

// ---------------------------------------------------------------------------
// LayerNorm over rows of length C=1024 (one block per row, 256 threads x 4)
// ---------------------------------------------------------------------------
template <bool WRITE_BF16>
__global__ __launch_bounds__(256) void layernorm_kernel(
    const float* __restrict__ in, const float* __restrict__ gamma,
    const float* __restrict__ beta, float* __restrict__ outf,
    __bf16* __restrict__ outb) {
  __shared__ float red[256];
  const int row = blockIdx.x;
  const int tid = threadIdx.x;
  const float4 x4 = ((const float4*)(in + (size_t)row * C_EMB))[tid];

  float s = x4.x + x4.y + x4.z + x4.w;
  red[tid] = s;
  __syncthreads();
  for (int st = 128; st > 0; st >>= 1) {
    if (tid < st) red[tid] += red[tid + st];
    __syncthreads();
  }
  const float mu = red[0] * (1.0f / C_EMB);
  __syncthreads();

  float dx = x4.x - mu, dy = x4.y - mu, dz = x4.z - mu, dw = x4.w - mu;
  red[tid] = dx * dx + dy * dy + dz * dz + dw * dw;
  __syncthreads();
  for (int st = 128; st > 0; st >>= 1) {
    if (tid < st) red[tid] += red[tid + st];
    __syncthreads();
  }
  const float rs = rsqrtf(red[0] * (1.0f / C_EMB) + 1e-5f);

  const int j = tid * 4;
  const float y0 = dx * rs * gamma[j]     + beta[j];
  const float y1 = dy * rs * gamma[j + 1] + beta[j + 1];
  const float y2 = dz * rs * gamma[j + 2] + beta[j + 2];
  const float y3 = dw * rs * gamma[j + 3] + beta[j + 3];
  float4 y; y.x = y0; y.y = y1; y.z = y2; y.w = y3;
  ((float4*)(outf + (size_t)row * C_EMB))[tid] = y;
  if (WRITE_BF16) {
    __bf16* ob = outb + (size_t)row * C_EMB + j;
    ob[0] = (__bf16)y0; ob[1] = (__bf16)y1; ob[2] = (__bf16)y2; ob[3] = (__bf16)y3;
  }
}

// ---------------------------------------------------------------------------
// Host-side launch
// ---------------------------------------------------------------------------
extern "C" void kernel_launch(void* const* d_in, const int* in_sizes, int n_in,
                              void* d_out, int out_size, void* d_ws, size_t ws_size,
                              hipStream_t stream) {
  const float* x     = (const float*)d_in[0];
  const float* Wq    = (const float*)d_in[1];
  const float* Wk    = (const float*)d_in[2];
  const float* Wv    = (const float*)d_in[3];
  const float* Wo    = (const float*)d_in[4];
  const float* bo    = (const float*)d_in[5];
  const float* g1    = (const float*)d_in[6];
  const float* b1    = (const float*)d_in[7];
  const float* W_ff1 = (const float*)d_in[8];
  const float* b_ff1 = (const float*)d_in[9];
  const float* W_ff2 = (const float*)d_in[10];
  const float* b_ff2 = (const float*)d_in[11];
  const float* g2    = (const float*)d_in[12];
  const float* b2    = (const float*)d_in[13];

  char* ws = (char*)d_ws;
  const size_t MB = 1ull << 20;
  __bf16* wqT  = (__bf16*)(ws + 0 * MB);    // 2 MB each
  __bf16* wkT  = (__bf16*)(ws + 2 * MB);
  __bf16* wvT  = (__bf16*)(ws + 4 * MB);
  __bf16* woT  = (__bf16*)(ws + 6 * MB);
  __bf16* wf1T = (__bf16*)(ws + 8 * MB);    // 8 MB (4096 x 1024)
  __bf16* wf2T = (__bf16*)(ws + 16 * MB);   // 8 MB (1024 x 4096)
  __bf16* xb   = (__bf16*)(ws + 24 * MB);   // 8 MB
  __bf16* qb   = (__bf16*)(ws + 32 * MB);   // 8 MB
  __bf16* kb   = (__bf16*)(ws + 40 * MB);   // 8 MB
  __bf16* vT   = (__bf16*)(ws + 48 * MB);   // 8 MB
  __bf16* ctxb = (__bf16*)(ws + 56 * MB);   // 8 MB
  float*  res1 = (float*)(ws + 64 * MB);    // 16 MB
  float*  hf   = (float*)(ws + 80 * MB);    // 16 MB
  __bf16* hb   = (__bf16*)(ws + 96 * MB);   // 8 MB
  __bf16* ff1b = (__bf16*)(ws + 32 * MB);   // 32 MB, reuses qb/kb/vT/ctxb region
  float*  res2 = (float*)(ws + 64 * MB);    // reuses res1

  const dim3 pwBlk(32, 8);
  pack_wT_kernel<<<dim3(C_EMB / 32, C_EMB / 32), pwBlk, 0, stream>>>(Wq, wqT, C_EMB, C_EMB);
  pack_wT_kernel<<<dim3(C_EMB / 32, C_EMB / 32), pwBlk, 0, stream>>>(Wk, wkT, C_EMB, C_EMB);
  pack_wT_kernel<<<dim3(C_EMB / 32, C_EMB / 32), pwBlk, 0, stream>>>(Wv, wvT, C_EMB, C_EMB);
  pack_wT_kernel<<<dim3(C_EMB / 32, C_EMB / 32), pwBlk, 0, stream>>>(Wo, woT, C_EMB, C_EMB);
  pack_wT_kernel<<<dim3(FFDIM / 32, C_EMB / 32), pwBlk, 0, stream>>>(W_ff1, wf1T, C_EMB, FFDIM);
  pack_wT_kernel<<<dim3(C_EMB / 32, FFDIM / 32), pwBlk, 0, stream>>>(W_ff2, wf2T, FFDIM, C_EMB);
  pack_x_kernel<<<(MROWS * C_EMB) / 1024, 1024, 0, stream>>>(x, xb, MROWS * C_EMB);

  // QKV projections
  gemm_bf16_kernel<0><<<dim3(C_EMB / 128, MROWS / 128), 128, 0, stream>>>(
      xb, wqT, MROWS, C_EMB, C_EMB, nullptr, qb, nullptr, nullptr);
  gemm_bf16_kernel<0><<<dim3(C_EMB / 128, MROWS / 128), 128, 0, stream>>>(
      xb, wkT, MROWS, C_EMB, C_EMB, nullptr, kb, nullptr, nullptr);
  gemm_bf16_kernel<1><<<dim3(C_EMB / 128, MROWS / 128), 128, 0, stream>>>(
      xb, wvT, MROWS, C_EMB, C_EMB, nullptr, vT, nullptr, nullptr);

  // Causal flash attention
  attn_kernel<<<dim3(T_SEQ / 16, 2 * HEADS), 32, 0, stream>>>(qb, kb, vT, ctxb);

  // Output projection + bias + residual(x)
  gemm_bf16_kernel<2><<<dim3(C_EMB / 128, MROWS / 128), 128, 0, stream>>>(
      ctxb, woT, MROWS, C_EMB, C_EMB, res1, nullptr, bo, x);
  layernorm_kernel<true><<<MROWS, 256, 0, stream>>>(res1, g1, b1, hf, hb);

  // FFN
  gemm_bf16_kernel<3><<<dim3(FFDIM / 128, MROWS / 128), 128, 0, stream>>>(
      hb, wf1T, MROWS, FFDIM, C_EMB, nullptr, ff1b, b_ff1, nullptr);
  gemm_bf16_kernel<2><<<dim3(C_EMB / 128, MROWS / 128), 128, 0, stream>>>(
      ff1b, wf2T, MROWS, C_EMB, FFDIM, res2, nullptr, b_ff2, hf);
  layernorm_kernel<false><<<MROWS, 256, 0, stream>>>(res2, g2, b2, (float*)d_out, nullptr);
}
